// BertSelfAttention_71055938945073
// MI455X (gfx1250) — matrix operand order repaired and checked by reading, not compiled, not run on hardware
//
#include <hip/hip_runtime.h>
#include <hip/hip_bf16.h>
#include <stdint.h>

// ---------------------------------------------------------------------------
// BERT self-attention (relative_key_query) for MI455X / gfx1250, wave32 WMMA.
// B=4, S=1024, H=1024, NH=16, HD=64, MAX_POS=1024.
// ---------------------------------------------------------------------------

#define S_LEN 1024
#define HID   1024
#define HDIM  64
#define NHEAD 16

typedef __attribute__((ext_vector_type(16))) __bf16 v16bf;
typedef __attribute__((ext_vector_type(8)))  float  v8f;
typedef __attribute__((ext_vector_type(4)))  unsigned int v4u;
typedef __attribute__((ext_vector_type(8)))  int    v8i;
typedef __attribute__((ext_vector_type(4)))  int    v4i;

union BFrag { uint4 u[2]; v16bf v; };

__device__ __forceinline__ unsigned short f2bf(float f) {
  unsigned u = __float_as_uint(f);
  u += 0x7FFFu + ((u >> 16) & 1u);            // round-to-nearest-even
  return (unsigned short)(u >> 16);
}

__device__ __forceinline__ v8f wmma_bf16(const BFrag& a, const BFrag& b, v8f c) {
  return __builtin_amdgcn_wmma_f32_16x16x32_bf16(false, a.v, false, b.v,
                                                 (short)0, c, false, false);
}

// ---- Tensor Data Mover: 2D tile (row_bytes x rows) global -> LDS ----------
// D# per CDNA5 ISA: data_size=1B, tensor==tile dims,
// type=2, count=1, trailing groups zero (tile_dim3/4 = 0 => unused).
// This toolchain's builtin takes 6 args: (v4u, v8i, v4i, v4i, v8i, i32 cpol).
__device__ __forceinline__ void tdm_load_2d(unsigned lds_addr, const void* gptr,
                                            unsigned row_bytes, unsigned rows,
                                            unsigned stride_bytes) {
  unsigned long long ga = (unsigned long long)(uintptr_t)gptr;
  v4u g0;
  g0.x = 1u;                                     // count=1, user mode
  g0.y = lds_addr;                               // LDS byte address
  g0.z = (unsigned)ga;                           // global_addr[31:0]
  g0.w = (unsigned)((ga >> 32) & 0x01FFFFFFu) | 0x80000000u;  // addr[56:32] | type=2
  v8i g1;
  g1[0] = 0;                                     // wg_mask=0, data_size=0 (1B)
  g1[1] = (int)(row_bytes << 16);                // tensor_dim0[15:0] @ bits63:48
  g1[2] = (int)((row_bytes >> 16) | (rows << 16));   // dim0 hi | tensor_dim1 lo
  g1[3] = (int)((rows >> 16) | (row_bytes << 16));   // dim1 hi | tile_dim0
  g1[4] = (int)(rows & 0xFFFFu);                 // tile_dim1 (tile_dim2 = 0)
  g1[5] = (int)stride_bytes;                     // tensor_dim0_stride[31:0]
  g1[6] = 0;
  g1[7] = 0;
  v4i z4 = {0, 0, 0, 0};
  v8i z8 = {0, 0, 0, 0, 0, 0, 0, 0};
  __builtin_amdgcn_tensor_load_to_lds(g0, g1, z4, z4, z8, 0);
}

// ---------------------------------------------------------------------------
// Kernel 0: f32 -> bf16 cast (with zero-pad tail for dist_emb padding row)
// ---------------------------------------------------------------------------
__global__ void cvt_f32_bf16(const float* __restrict__ src,
                             unsigned short* __restrict__ dst,
                             int n_src, int n_dst) {
  int stride = gridDim.x * blockDim.x;
  for (int i = blockIdx.x * blockDim.x + threadIdx.x; i < n_dst; i += stride)
    dst[i] = (i < n_src) ? f2bf(src[i]) : (unsigned short)0;
}

// ---------------------------------------------------------------------------
// Kernel 1: out = X @ W^T + bias, bf16 WMMA, f32 accumulate, bf16 store.
// X [4096,1024], W [1024,1024] row-major (k contiguous). 4 waves/block,
// each wave owns a 16x64 output strip of a 64x64 block tile.
// transposeV: store out[(b*1024+n)*1024 + s] instead of out[m*1024 + n].
// ---------------------------------------------------------------------------
__global__ __launch_bounds__(128)
void qkv_gemm(const unsigned short* __restrict__ X,
              const unsigned short* __restrict__ W,
              const float* __restrict__ bias,
              unsigned short* __restrict__ out,
              int transposeV) {
  const int lane = threadIdx.x & 31;
  const int w    = threadIdx.x >> 5;
  const int hi   = lane >> 4;
  const int lw   = lane & 15;
  const int n0   = blockIdx.x * 64;
  const int m0   = blockIdx.y * 64;

  const unsigned short* arow = X + (size_t)(m0 + w * 16 + lw) * HID;
  v8f acc[4] = {};

  #pragma unroll 4
  for (int k0 = 0; k0 < HID; k0 += 32) {
    BFrag a;
    const int ab = k0 + hi * 16;
    a.u[0] = *(const uint4*)(arow + ab);
    a.u[1] = *(const uint4*)(arow + ab + 8);
    #pragma unroll
    for (int nt = 0; nt < 4; ++nt) {
      BFrag bm;
      const unsigned short* brow =
          W + (size_t)(n0 + nt * 16 + lw) * HID + k0 + hi * 16;
      bm.u[0] = *(const uint4*)(brow);
      bm.u[1] = *(const uint4*)(brow + 8);
      acc[nt] = wmma_bf16(a, bm, acc[nt]);
    }
  }

  const int b = (m0 + w * 16 + lw) >> 10;    // batch index (tile never crosses)
  #pragma unroll
  for (int nt = 0; nt < 4; ++nt) {
    const int n = n0 + nt * 16 + lw;
    const float bv = bias[n];
    #pragma unroll
    for (int v = 0; v < 8; ++v) {
      const int m = m0 + w * 16 + v + 8 * hi;
      const unsigned short val = f2bf(acc[nt][v] + bv);
      if (!transposeV) out[(size_t)m * HID + n] = val;
      else             out[((size_t)((b << 10) + n) << 10) + (m & 1023)] = val;
    }
  }
}

// ---------------------------------------------------------------------------
// Kernel 2: fused flash attention with banded relative-position bias.
// Block = 128 threads (4 waves), 64 queries of one (b,h). KV step N=64.
//  scores[l,r] = (q·k + q·D[l-r+1023] + k·D[l-r+1023]) / 8 + mask[b,r]
// QD band: wave w computes [16 q-rows] x [c in 16w..16w+79]  (c=(l-l0)-(r-r0)+63)
// KD band: wave w computes [16 k-rows] x [c in 48-16w..127-16w]
// ---------------------------------------------------------------------------
__global__ __launch_bounds__(128)
void attention(const unsigned short* __restrict__ Q,   // [4096,1024] bf16
               const unsigned short* __restrict__ K,   // [4096,1024] bf16
               const unsigned short* __restrict__ Vt,  // [(b*1024+h*64+d)][s]
               const unsigned short* __restrict__ D,   // [2048,64] bf16 (padded)
               const float* __restrict__ mask,         // [4,1024]
               float* __restrict__ out) {
  __shared__ __align__(16) unsigned char smem[57344];
  float*          QDs = (float*)smem;                       // [4][16][80]
  float*          KDs = QDs + 4 * 16 * 80;                  // [4][16][80]
  unsigned short* KPs = (unsigned short*)(smem + 40960);    // [64][64] K, then P
  unsigned short* Vts = KPs + 64 * 64;                      // [64][64] V^T tile

  const int lane = threadIdx.x & 31;
  const int w    = threadIdx.x >> 5;
  const int hi   = lane >> 4;
  const int lw   = lane & 15;
  const int l0   = blockIdx.x * 64;
  const int bh   = blockIdx.y;
  const int b    = bh >> 4, h = bh & 15;

  // Q A-fragments for this wave's 16 rows, kept in registers for all GEMMs.
  BFrag qf[2];
  {
    const unsigned short* qrow =
        Q + (size_t)(b * S_LEN + l0 + w * 16 + lw) * HID + h * HDIM;
    #pragma unroll
    for (int ks = 0; ks < 2; ++ks) {
      const int base = ks * 32 + hi * 16;
      qf[ks].u[0] = *(const uint4*)(qrow + base);
      qf[ks].u[1] = *(const uint4*)(qrow + base + 8);
    }
  }

  v8f   acc_o[4] = {};
  float mrow[8], lrow[8];
  #pragma unroll
  for (int v = 0; v < 8; ++v) { mrow[v] = -1e30f; lrow[v] = 0.f; }

  for (int r0 = 0; r0 < S_LEN; r0 += 64) {
    __syncthreads();                         // prior LDS consumers done
    if (w == 0) {
      // TDM DMA: K tile [64 rows x 128B] and V^T tile [64 rows x 128B] -> LDS
      tdm_load_2d((unsigned)(uintptr_t)KPs,
                  K + (size_t)(b * S_LEN + r0) * HID + h * HDIM,
                  128, 64, HID * 2);
      tdm_load_2d((unsigned)(uintptr_t)Vts,
                  Vt + (size_t)(b * S_LEN + h * HDIM) * S_LEN + r0,
                  128, 64, S_LEN * 2);
      __builtin_amdgcn_s_wait_tensorcnt(0);
    }
    __syncthreads();

    const int jb = l0 - r0 + 960;            // dist row = jb + c, c in [0,127]

    // ---- S = Q K^T (4 col-tiles) ----
    v8f sacc[4] = {};
    #pragma unroll
    for (int ks = 0; ks < 2; ++ks) {
      #pragma unroll
      for (int nt = 0; nt < 4; ++nt) {
        BFrag bm;
        const unsigned short* kr = KPs + (nt * 16 + lw) * 64 + ks * 32 + hi * 16;
        bm.u[0] = *(const uint4*)(kr);
        bm.u[1] = *(const uint4*)(kr + 8);
        sacc[nt] = wmma_bf16(qf[ks], bm, sacc[nt]);
      }
    }

    // ---- QD band: q(16) x D^T(80 cols starting at c0 = 16w) ----
    #pragma unroll
    for (int t = 0; t < 5; ++t) {
      const int c0 = w * 16 + t * 16;
      v8f acc = {};
      #pragma unroll
      for (int ks = 0; ks < 2; ++ks) {
        BFrag bm;
        const unsigned short* dr =
            D + (size_t)(jb + c0 + lw) * HDIM + ks * 32 + hi * 16;
        bm.u[0] = *(const uint4*)(dr);
        bm.u[1] = *(const uint4*)(dr + 8);
        acc = wmma_bf16(qf[ks], bm, acc);
      }
      #pragma unroll
      for (int v = 0; v < 8; ++v)
        QDs[(w * 16 + v + 8 * hi) * 80 + t * 16 + lw] = acc[v];
    }

    // ---- KD band: k-rows [16w..16w+15] x D^T(80 cols at c0 = 48-16w) ----
    BFrag kf[2];
    #pragma unroll
    for (int ks = 0; ks < 2; ++ks) {
      const unsigned short* kr = KPs + (w * 16 + lw) * 64 + ks * 32 + hi * 16;
      kf[ks].u[0] = *(const uint4*)(kr);
      kf[ks].u[1] = *(const uint4*)(kr + 8);
    }
    #pragma unroll
    for (int t = 0; t < 5; ++t) {
      const int c0 = 48 - w * 16 + t * 16;
      v8f acc = {};
      #pragma unroll
      for (int ks = 0; ks < 2; ++ks) {
        BFrag bm;
        const unsigned short* dr =
            D + (size_t)(jb + c0 + lw) * HDIM + ks * 32 + hi * 16;
        bm.u[0] = *(const uint4*)(dr);
        bm.u[1] = *(const uint4*)(dr + 8);
        acc = wmma_bf16(kf[ks], bm, acc);
      }
      #pragma unroll
      for (int v = 0; v < 8; ++v)
        KDs[(w * 16 + v + 8 * hi) * 80 + t * 16 + lw] = acc[v];
    }
    __syncthreads();                         // bands visible, K reads done

    // ---- gather diagonals, scale, mask, online softmax ----
    float p[4][8], tmax[8];
    #pragma unroll
    for (int v = 0; v < 8; ++v) tmax[v] = -1e30f;
    #pragma unroll
    for (int nt = 0; nt < 4; ++nt) {
      const float mk = mask[b * S_LEN + r0 + nt * 16 + lw];
      #pragma unroll
      for (int v = 0; v < 8; ++v) {
        const int m   = v + 8 * hi;
        const int clq = m - (nt * 16 + lw) + 63;      // [0,78]
        const int clk = 16 * w + m - lw + 15;         // [0,78]
        const float qd = QDs[(w * 16 + m) * 80 + clq];
        const float kd = KDs[(nt * 16 + lw) * 80 + clk];
        const float s  = (sacc[nt][v] + qd + kd) * 0.125f + mk;
        p[nt][v] = s;
        tmax[v]  = fmaxf(tmax[v], s);
      }
    }
    #pragma unroll
    for (int v = 0; v < 8; ++v) {
      float t = tmax[v];
      t = fmaxf(t, __shfl_xor(t, 1, 32));
      t = fmaxf(t, __shfl_xor(t, 2, 32));
      t = fmaxf(t, __shfl_xor(t, 4, 32));
      t = fmaxf(t, __shfl_xor(t, 8, 32));
      const float mnew  = fmaxf(mrow[v], t);
      const float alpha = __expf(mrow[v] - mnew);
      mrow[v] = mnew;
      float rs = 0.f;
      #pragma unroll
      for (int nt = 0; nt < 4; ++nt) {
        const float e = __expf(p[nt][v] - mnew);
        p[nt][v] = e;
        rs += e;
      }
      rs += __shfl_xor(rs, 1, 32);
      rs += __shfl_xor(rs, 2, 32);
      rs += __shfl_xor(rs, 4, 32);
      rs += __shfl_xor(rs, 8, 32);
      lrow[v] = lrow[v] * alpha + rs;
      #pragma unroll
      for (int dt = 0; dt < 4; ++dt) acc_o[dt][v] = acc_o[dt][v] * alpha;
    }

    // ---- P -> LDS (overlays K tile) in A-feedable row-major ----
    #pragma unroll
    for (int nt = 0; nt < 4; ++nt)
      #pragma unroll
      for (int v = 0; v < 8; ++v)
        KPs[(w * 16 + v + 8 * hi) * 64 + nt * 16 + lw] = f2bf(p[nt][v]);
    __syncthreads();

    // ---- acc_o += P V ----
    #pragma unroll
    for (int ks = 0; ks < 2; ++ks) {
      BFrag pf;
      const unsigned short* pr = KPs + (w * 16 + lw) * 64 + ks * 32 + hi * 16;
      pf.u[0] = *(const uint4*)(pr);
      pf.u[1] = *(const uint4*)(pr + 8);
      #pragma unroll
      for (int dt = 0; dt < 4; ++dt) {
        BFrag bm;
        const unsigned short* vr = Vts + (dt * 16 + lw) * 64 + ks * 32 + hi * 16;
        bm.u[0] = *(const uint4*)(vr);
        bm.u[1] = *(const uint4*)(vr + 8);
        acc_o[dt] = wmma_bf16(pf, bm, acc_o[dt]);
      }
    }
  }

  // ---- epilogue: out[b][s][h*64+d] = acc / l ----
  #pragma unroll
  for (int dt = 0; dt < 4; ++dt)
    #pragma unroll
    for (int v = 0; v < 8; ++v) {
      const int l = l0 + w * 16 + v + 8 * hi;
      out[(size_t)(b * S_LEN + l) * HID + h * HDIM + dt * 16 + lw] =
          acc_o[dt][v] / lrow[v];
    }
}

// ---------------------------------------------------------------------------
extern "C" void kernel_launch(void* const* d_in, const int* in_sizes, int n_in,
                              void* d_out, int out_size, void* d_ws, size_t ws_size,
                              hipStream_t stream) {
  const float* hidden = (const float*)d_in[0];
  const float* amask  = (const float*)d_in[1];
  const float* wq     = (const float*)d_in[2];
  const float* bq     = (const float*)d_in[3];
  const float* wk     = (const float*)d_in[4];
  const float* bk     = (const float*)d_in[5];
  const float* wv     = (const float*)d_in[6];
  const float* bv     = (const float*)d_in[7];
  const float* dist   = (const float*)d_in[8];

  // workspace layout (bf16 halves), total ~38.3 MB
  unsigned short* X16  = (unsigned short*)d_ws;       // 4096*1024
  unsigned short* Wq16 = X16  + 4096 * 1024;          // 1024*1024
  unsigned short* Wk16 = Wq16 + 1024 * 1024;
  unsigned short* Wv16 = Wk16 + 1024 * 1024;
  unsigned short* D16  = Wv16 + 1024 * 1024;          // 2048*64 (row 2047 = 0)
  unsigned short* Q16  = D16  + 2048 * 64;            // 4096*1024
  unsigned short* K16  = Q16  + 4096 * 1024;
  unsigned short* V16t = K16  + 4096 * 1024;

  cvt_f32_bf16<<<4096, 256, 0, stream>>>(hidden, X16, 4096 * 1024, 4096 * 1024);
  cvt_f32_bf16<<<2048, 256, 0, stream>>>(wq, Wq16, 1024 * 1024, 1024 * 1024);
  cvt_f32_bf16<<<2048, 256, 0, stream>>>(wk, Wk16, 1024 * 1024, 1024 * 1024);
  cvt_f32_bf16<<<2048, 256, 0, stream>>>(wv, Wv16, 1024 * 1024, 1024 * 1024);
  cvt_f32_bf16<<<512, 256, 0, stream>>>(dist, D16, 2047 * 64, 2048 * 64);

  dim3 gg(HID / 64, (4 * S_LEN) / 64);
  qkv_gemm<<<gg, 128, 0, stream>>>(X16, Wq16, bq, Q16, 0);
  qkv_gemm<<<gg, 128, 0, stream>>>(X16, Wk16, bk, K16, 0);
  qkv_gemm<<<gg, 128, 0, stream>>>(X16, Wv16, bv, V16t, 1);

  attention<<<dim3(S_LEN / 64, 4 * NHEAD), 128, 0, stream>>>(
      Q16, K16, V16t, D16, amask, (float*)d_out);
}
